// RNNSeq2Seq_68436008894915
// MI455X (gfx1250) — compile-verified, matrix operand order checked by reading, not run on hardware
//
#include <hip/hip_runtime.h>
#include <math.h>
#include <stdint.h>

#define HD    512
#define GD    1536
#define BATCH 32
#define TSEQ  256
#define VOC   50257
#define NTILES_V 3142            // ceil(50257/16)

typedef __attribute__((ext_vector_type(16))) __bf16 v16bf;
typedef __attribute__((ext_vector_type(8)))  float  v8f;
typedef unsigned int u32x4 __attribute__((ext_vector_type(4)));
typedef int i32x4 __attribute__((ext_vector_type(4)));
typedef int i32x8 __attribute__((ext_vector_type(8)));

#if defined(__has_builtin)
#if __has_builtin(__builtin_amdgcn_tensor_load_to_lds)
#define HAVE_TDM 1
#endif
#endif
#ifndef HAVE_TDM
#define HAVE_TDM 0
#endif

union Frag {
    v16bf v;
    unsigned short u[16];
    unsigned int   d[8];
};

static __device__ inline unsigned short f2bf(float f) {
    unsigned int x = __float_as_uint(f);
    unsigned int r = x + 0x7FFFu + ((x >> 16) & 1u);   // round-to-nearest-even
    return (unsigned short)(r >> 16);
}

// K index (pair base, even) inside a 32-wide K tile for bf16 A/B fragments.
// Lanes 0-15 hold K {0..7, 16..23}; lanes 16-31 hold K {8..15, 24..31}. (ISA 7.12.2)
static __device__ inline int bfragK(int v, int lane) {
    int b = (v < 4) ? (2 * v) : (16 + 2 * (v - 4));
    return b + ((lane & 16) ? 8 : 0);
}

static __device__ inline v8f wmma_bf16(v16bf a, v16bf b, v8f c) {
    return __builtin_amdgcn_wmma_f32_16x16x32_bf16(false, a, false, b, (short)0, c,
                                                   false, false);
}

static __device__ inline float sigf(float x) {
    return 1.0f / (1.0f + __expf(-x));
}

// ---------------- device-wide barrier (32 resident WGs) ----------------
static __device__ inline void grid_barrier(int* cnt, int* gen, int nwg) {
    __syncthreads();
    if (threadIdx.x == 0) {
        __threadfence();
        int g = __hip_atomic_load(gen, __ATOMIC_RELAXED, __HIP_MEMORY_SCOPE_AGENT);
        int p = __hip_atomic_fetch_add(cnt, 1, __ATOMIC_ACQ_REL, __HIP_MEMORY_SCOPE_AGENT);
        if (p == nwg - 1) {
            __hip_atomic_store(cnt, 0, __ATOMIC_RELAXED, __HIP_MEMORY_SCOPE_AGENT);
            __hip_atomic_fetch_add(gen, 1, __ATOMIC_RELEASE, __HIP_MEMORY_SCOPE_AGENT);
        } else {
            while (__hip_atomic_load(gen, __ATOMIC_ACQUIRE, __HIP_MEMORY_SCOPE_AGENT) == g)
                __builtin_amdgcn_s_sleep(2);
        }
        __threadfence();
    }
    __syncthreads();
}

// ---------------- init: zero hidden state + sync cells ----------------
__global__ void init_state(float* __restrict__ h, int* __restrict__ syn) {
    int i = blockIdx.x * 256 + threadIdx.x;
    if (i < BATCH * HD) h[i] = 0.0f;
    if (i < 2) syn[i] = 0;
}

// ---------------- embedding gather ----------------
__global__ void embed_gather(const int* __restrict__ tok, const float* __restrict__ emb,
                             float* __restrict__ out, int n /*B*T*/) {
    int i = blockIdx.x * 256 + threadIdx.x;
    if (i >= n * HD) return;
    int bt = i >> 9;
    int hh = i & (HD - 1);
    out[i] = emb[(size_t)tok[bt] * HD + hh];
}

// ---------------- pack h (f32 [32,512]) into bf16 A-fragment layout ----------------
// hpack layout: [kt=16][mt=2][lane=32][16 halves]  (parity 0 buffer)
__global__ void pack_state(const float* __restrict__ h, unsigned short* __restrict__ hpack) {
    int e = blockIdx.x * 256 + threadIdx.x;
    if (e >= 16 * 2 * 32) return;
    int lane = e & 31, mt = (e >> 5) & 1, kt = e >> 6;
    Frag f;
    int m = mt * 16 + (lane & 15);
#pragma unroll
    for (int v = 0; v < 8; ++v) {
        int k = kt * 32 + bfragK(v, lane);
        f.u[2 * v]     = f2bf(h[m * HD + k]);
        f.u[2 * v + 1] = f2bf(h[m * HD + k + 1]);
    }
    *(v16bf*)(hpack + (size_t)e * 16) = f.v;
}

// ---------------- pack a [512,N] f32 B-matrix into bf16 B-fragments ----------------
// out layout: [kt=16][ntile=N/16][lane=32][16 halves]
__global__ void pack_bfrag(const float* __restrict__ W, int N,
                           unsigned short* __restrict__ out) {
    int i = blockIdx.x * 256 + threadIdx.x;     // one lane-fragment per thread
    int lane = i & 31;
    int rest = i >> 5;
    int ntiles = N >> 4;
    int nt = rest % ntiles, kt = rest / ntiles;
    if (kt >= 16) return;
    int col = nt * 16 + (lane & 15);
    Frag f;
#pragma unroll
    for (int v = 0; v < 8; ++v) {
        int k = kt * 32 + bfragK(v, lane);
        f.u[2 * v]     = f2bf(W[(size_t)k * N + col]);
        f.u[2 * v + 1] = f2bf(W[(size_t)(k + 1) * N + col]);
    }
    *(v16bf*)(out + (size_t)i * 16) = f.v;
}

// ---------------- pack X [8192,512] f32 into bf16 A-fragments ----------------
// out layout: [mtile=512][kt=16][lane=32][16 halves]
__global__ void pack_x(const float* __restrict__ X, unsigned short* __restrict__ out) {
    int i = blockIdx.x * 256 + threadIdx.x;     // 512*16*32 lane-fragments
    int lane = i & 31;
    int rest = i >> 5;
    int kt = rest & 15, mtile = rest >> 4;
    int m = mtile * 16 + (lane & 15);
    Frag f;
#pragma unroll
    for (int v = 0; v < 8; ++v) {
        int k = kt * 32 + bfragK(v, lane);
        f.u[2 * v]     = f2bf(X[(size_t)m * HD + k]);
        f.u[2 * v + 1] = f2bf(X[(size_t)m * HD + k + 1]);
    }
    *(v16bf*)(out + (size_t)i * 16) = f.v;
}

// ---------------- big GEMM: C[8192,1536] = X*W from pre-packed fragments -------------
// No LDS, no conversion: contiguous 32B/lane fragment loads + WMMA only.
__global__ __launch_bounds__(256) void xproj_gemm(const unsigned short* __restrict__ Apk,
                                                  const unsigned short* __restrict__ Bpk,
                                                  float* __restrict__ C) {
    const int tid = threadIdx.x, lane = tid & 31, wave = tid >> 5;
    const int mt = wave & 3, nh = wave >> 2;     // wave -> (Mtile, Nhalf)
    const int mtile0 = blockIdx.x * 4;           // 64 rows per WG
    const int nt0 = blockIdx.y * 8;              // 128 cols per WG
    v8f acc[4] = {};
    for (int kt = 0; kt < 16; ++kt) {
        Frag a;
        a.v = *(const v16bf*)(Apk + (size_t)((mtile0 + mt) * 16 + kt) * 512 + lane * 16);
#pragma unroll
        for (int j = 0; j < 4; ++j) {
            Frag b;
            b.v = *(const v16bf*)(Bpk + (size_t)(kt * 96 + nt0 + nh * 4 + j) * 512 +
                                  lane * 16);
            acc[j] = wmma_bf16(a.v, b.v, acc[j]);
        }
    }
    const int hi8 = (lane & 16) ? 8 : 0;
#pragma unroll
    for (int j = 0; j < 4; ++j) {
        int n = (nt0 + nh * 4 + j) * 16 + (lane & 15);
#pragma unroll
        for (int v = 0; v < 8; ++v) {
            int m = (mtile0 + mt) * 16 + hi8 + v;
            C[(size_t)m * GD + n] = acc[j][v];
        }
    }
}

// ---------------- GRU recurrent scan ----------------
// 32 WGs x 192 threads (6 waves). WG w owns hidden units [16w,16w+16).
// Its 3 gate column strips of pre-packed Wh fragments (48KB) are copied into LDS
// once via the Tensor Data Mover (3 strided 2-D tiles), then reused for all steps.
__global__ __launch_bounds__(192) void gru_scan(
    const float* __restrict__ xproj,            // [B*T, 1536]
    const unsigned short* __restrict__ whg,     // packed Wh [kt][96][lane][16]
    const float* __restrict__ bias,             // [2, 1536]  (bx, bh)
    float* __restrict__ Y,                      // [B*T, 512]
    float* __restrict__ hstate,                 // [32, 512]
    unsigned short* __restrict__ hpack,         // [2 par][16 kt][2 mt][32][16]
    int T, int* sync_cnt, int* sync_gen) {
    __shared__ __align__(32) unsigned short whp[3 * 16 * 32 * 16];  // [g][kt][lane][16]
    __shared__ __align__(32) float red[2 * 2 * 32 * 8];             // r/h gate partials
    __shared__ __align__(4) unsigned short hloc[32 * 16];           // new h (bf16)
    const int w = blockIdx.x;
    const int tid = threadIdx.x, lane = tid & 31, wave = tid >> 5;
    const int g = wave >> 1, mt = wave & 1;

#if HAVE_TDM
    if (wave == 0) {
        // One 2-D TDM tile per gate: 16 rows (kt) x 1024B, row stride 96KB.
        for (int gg = 0; gg < 3; ++gg) {
            unsigned long long ga = (unsigned long long)(uintptr_t)whg +
                                    (unsigned long long)(gg * 32 + w) * 1024ull;
            unsigned ldsoff = (unsigned)(uintptr_t)(&whp[0]) + (unsigned)gg * 16384u;
            u32x4 g0 = {1u,                                   // count=1 valid descriptor
                        ldsoff,                               // lds_addr
                        (unsigned)ga,                         // global_addr[31:0]
                        (unsigned)((ga >> 32) & 0x1FFFFFFull) | (2u << 30)};  // type=2
            i32x8 g1 = {(int)(3u << 16),      // data_size = 8 bytes
                        (int)(128u << 16),    // tensor_dim0 = 128 (bits 79:48)
                        (int)(16u << 16),     // tensor_dim1 = 16  (bits 111:80)
                        (int)(128u << 16),    // tile_dim0  = 128 (bits 127:112)
                        16,                   // tile_dim1  = 16
                        12288,                // tensor_dim0_stride = 96KB / 8B
                        0, 0};
            i32x4 gz = {0, 0, 0, 0};
#if __clang_major__ >= 23
            i32x8 gz8 = {0, 0, 0, 0, 0, 0, 0, 0};
            __builtin_amdgcn_tensor_load_to_lds(g0, g1, gz, gz, gz8, 0);
#else
            __builtin_amdgcn_tensor_load_to_lds(g0, g1, gz, gz, 0);
#endif
        }
        __builtin_amdgcn_s_wait_tensorcnt(0);
    }
#else
    for (int i = tid; i < 12288; i += 192) {    // fallback: plain strided copy
        int d = i & 255;
        int blk = i >> 8;                       // g*16 + kt
        int gg = blk >> 4, kt = blk & 15;
        ((unsigned*)whp)[blk * 256 + d] =
            ((const unsigned*)whg)[(size_t)(kt * 96 + gg * 32 + w) * 256 + d];
    }
#endif
    __syncthreads();

    const int ul = lane & 15;
    const int u = w * 16 + ul;
    const int hi8 = (lane & 16) ? 8 : 0;
    const float bxz = bias[u],      bxr = bias[512 + u],      bxh = bias[1024 + u];
    const float bhz = bias[GD + u], bhr = bias[GD + 512 + u], bhh = bias[GD + 1024 + u];

    for (int t = 0; t < T; ++t) {
        const int par = t & 1;
        const unsigned short* hin = hpack + (size_t)par * 16384;
        v8f acc = {};
#pragma unroll
        for (int kt = 0; kt < 16; ++kt) {
            Frag a, b;
            a.v = *(const v16bf*)(hin + (size_t)(kt * 2 + mt) * 512 + lane * 16);
            b.v = *(const v16bf*)(whp + (size_t)((g * 16 + kt) * 32 + lane) * 16);
            acc = wmma_bf16(a.v, b.v, acc);
        }
        if (g != 0) {
            float* rp = &red[(size_t)(((g - 1) * 2 + mt) * 32 + lane) * 8];
#pragma unroll
            for (int v = 0; v < 8; ++v) rp[v] = acc[v];
        }
        __syncthreads();
        if (g == 0) {                           // fused gate update on z-waves
            const float* rr = &red[(size_t)((0 * 2 + mt) * 32 + lane) * 8];
            const float* rh = &red[(size_t)((1 * 2 + mt) * 32 + lane) * 8];
#pragma unroll
            for (int v = 0; v < 8; ++v) {
                int b = mt * 16 + hi8 + v;
                const float* xp = xproj + (size_t)(b * T + t) * GD;
                float z  = sigf(xp[u] + bxz + acc[v] + bhz);
                float r  = sigf(xp[512 + u] + bxr + rr[v] + bhr);
                float hh = tanhf(xp[1024 + u] + bxh + r * (rh[v] + bhh));
                float ho = hstate[b * HD + u];
                float hv = z * ho + (1.0f - z) * hh;
                Y[(size_t)(b * T + t) * HD + u] = hv;
                hstate[b * HD + u] = hv;
                hloc[b * 16 + ul] = f2bf(hv);
            }
        }
        __syncthreads();
        if (g == 0) {                           // repack new h into A-fragment layout
            unsigned short* dst = hpack + (size_t)(par ^ 1) * 16384 +
                                  (size_t)((w >> 1) * 2 + mt) * 512 + lane * 16;
            int m = mt * 16 + (lane & 15);
            int vbase = (w & 1) ? 4 : 0;        // this WG owns half of the 32-wide K tile
#pragma unroll
            for (int v2 = 0; v2 < 4; ++v2) {
                int v = vbase + v2;
                int kl = bfragK(v, lane) - ((w & 1) * 16);
                *(unsigned int*)(dst + 2 * v) = *(const unsigned int*)(hloc + m * 16 + kl);
            }
        }
        grid_barrier(sync_cnt, sync_gen, 32);
    }
}

// ---------------- dense head: out[32,50257] = h*W + b (A from packed hpk) ------------
__global__ __launch_bounds__(256) void dense_gemm(const unsigned short* __restrict__ hpk,
                                                  const float* __restrict__ W,
                                                  const float* __restrict__ bias,
                                                  float* __restrict__ out) {
    const int tid = threadIdx.x, lane = tid & 31, wave = tid >> 5;
    const int nt = blockIdx.x * 8 + wave;
    const int n = nt * 16 + (lane & 15);
    const bool nok = (n < VOC);
    v8f acc0 = {}, acc1 = {};
    for (int kt = 0; kt < 16; ++kt) {
        Frag bfg;
#pragma unroll
        for (int v = 0; v < 8; ++v) {
            int k = kt * 32 + bfragK(v, lane);
            float f0 = nok ? W[(size_t)k * VOC + n] : 0.0f;
            float f1 = nok ? W[(size_t)(k + 1) * VOC + n] : 0.0f;
            bfg.u[2 * v]     = f2bf(f0);
            bfg.u[2 * v + 1] = f2bf(f1);
        }
        if (kt < 15 && nok)
            __builtin_prefetch(&W[(size_t)(kt * 32 + 32) * VOC + n], 0, 1);
        Frag a0, a1;
        a0.v = *(const v16bf*)(hpk + (size_t)(kt * 2 + 0) * 512 + lane * 16);
        a1.v = *(const v16bf*)(hpk + (size_t)(kt * 2 + 1) * 512 + lane * 16);
        acc0 = wmma_bf16(a0.v, bfg.v, acc0);
        acc1 = wmma_bf16(a1.v, bfg.v, acc1);
    }
    if (nok) {
        float bb = bias[n];
        int hi8 = (lane & 16) ? 8 : 0;
#pragma unroll
        for (int v = 0; v < 8; ++v) {
            int b = hi8 + v;
            out[(size_t)b * VOC + n]        = acc0[v] + bb;
            out[(size_t)(b + 16) * VOC + n] = acc1[v] + bb;
        }
    }
}

// =====================================================================
extern "C" void kernel_launch(void* const* d_in, const int* in_sizes, int n_in,
                              void* d_out, int out_size, void* d_ws, size_t ws_size,
                              hipStream_t stream) {
    (void)in_sizes; (void)n_in; (void)out_size; (void)ws_size;
    const int*   enc_tok = (const int*)d_in[0];
    const int*   dec_tok = (const int*)d_in[1];
    const float* emb     = (const float*)d_in[2];
    const float* encWx   = (const float*)d_in[3];
    const float* encWh   = (const float*)d_in[4];
    const float* encB    = (const float*)d_in[5];
    const float* decWx   = (const float*)d_in[6];
    const float* decWh   = (const float*)d_in[7];
    const float* decB    = (const float*)d_in[8];
    const float* dW      = (const float*)d_in[9];
    const float* dB      = (const float*)d_in[10];
    float* outp = (float*)d_out;

    char* ws = (char*)d_ws;
    size_t off = 0;
    float* xbuf = (float*)(ws + off); off += (size_t)BATCH * TSEQ * HD * 4;     // 16MB
    float* xprj = (float*)(ws + off); off += (size_t)BATCH * TSEQ * GD * 4;     // 48MB
    unsigned short* xpk = (unsigned short*)(ws + off);
    off += (size_t)512 * 16 * 32 * 16 * 2;                                      // 8MB
    unsigned short* wxg = (unsigned short*)(ws + off);
    off += (size_t)16 * 96 * 32 * 16 * 2;                                       // 1.5MB
    unsigned short* whg = (unsigned short*)(ws + off);
    off += (size_t)16 * 96 * 32 * 16 * 2;                                       // 1.5MB
    float* hst  = (float*)(ws + off); off += (size_t)BATCH * HD * 4;            // 64KB
    unsigned short* hpk = (unsigned short*)(ws + off); off += 2 * 16384 * 2;    // 64KB
    int* syn = (int*)(ws + off); off += 256;

    init_state<<<64, 256, 0, stream>>>(hst, syn);

    embed_gather<<<(BATCH * TSEQ * HD) / 256, 256, 0, stream>>>(enc_tok, emb, xbuf,
                                                                BATCH * TSEQ);
    for (int l = 0; l < 2; ++l) {
        pack_bfrag<<<192, 256, 0, stream>>>(encWx + (size_t)l * HD * GD, GD, wxg);
        pack_bfrag<<<192, 256, 0, stream>>>(encWh + (size_t)l * HD * GD, GD, whg);
        pack_x<<<1024, 256, 0, stream>>>(xbuf, xpk);
        xproj_gemm<<<dim3(128, 12), 256, 0, stream>>>(xpk, wxg, xprj);
        pack_state<<<4, 256, 0, stream>>>(hst, hpk);
        gru_scan<<<32, 192, 0, stream>>>(xprj, whg, encB + (size_t)l * 2 * GD, xbuf,
                                         hst, hpk, TSEQ, syn, syn + 1);
    }
    embed_gather<<<(BATCH * TSEQ * HD) / 256, 256, 0, stream>>>(dec_tok, emb, xbuf,
                                                                BATCH * TSEQ);
    for (int l = 0; l < 2; ++l) {
        pack_bfrag<<<192, 256, 0, stream>>>(decWx + (size_t)l * HD * GD, GD, wxg);
        pack_bfrag<<<192, 256, 0, stream>>>(decWh + (size_t)l * HD * GD, GD, whg);
        pack_x<<<1024, 256, 0, stream>>>(xbuf, xpk);
        xproj_gemm<<<dim3(128, 12), 256, 0, stream>>>(xpk, wxg, xprj);
        pack_state<<<4, 256, 0, stream>>>(hst, hpk);
        gru_scan<<<32, 192, 0, stream>>>(xprj, whg, decB + (size_t)l * 2 * GD, xbuf,
                                         hst, hpk, TSEQ, syn, syn + 1);
    }
    pack_state<<<4, 256, 0, stream>>>(hst, hpk);
    dense_gemm<<<(NTILES_V + 7) / 8, 256, 0, stream>>>(hpk, dW, dB, outp);
}
